// CEABlock_Dynamic_27247272526359
// MI455X (gfx1250) — compile-verified, matrix operand order checked by reading, not run on hardware
//
#include <hip/hip_runtime.h>
#include <hip/hip_bf16.h>

// ---------------- CDNA5 WMMA types ----------------
typedef __attribute__((ext_vector_type(16))) __bf16       v16bf;
typedef __attribute__((ext_vector_type(8)))  float        v8f;
typedef __attribute__((ext_vector_type(8)))  unsigned int v8u;

// Problem constants
#define SB   32
#define LT   64
#define LS   256
#define LL   320      // LT + LS
#define CC   768
#define HH   12
#define HD   64
#define KEEPN 180     // ceil(0.7*256)
#define REMN  76
#define LNEW 244      // LT + KEEPN

// GEMM tiling
#define TM 128
#define TN 64
#define TK 32
#define ROWP 40       // padded LDS row (bf16 elems): 80B rows -> conflict-free frag loads

// ============================================================
// Generic batched GEMM:  C[m][n] (+)= sum_k A[m][k] * Bt[n][k]
// A: bf16 [M][K], Bt: bf16 [N][K], C: fp32/bf16 with ldc.
// Per-batch offsets: aOff=b*aBatch, bOff=b*bBatch,
//                    cOff=(b/cDiv)*cOuter + (b%cDiv)*cInner  (also for residual)
// Double-buffered LDS fed by global_load_async_to_lds_b128, software-pipelined
// one K-tile ahead (ASYNCcnt tracked; each wave issues exactly 3 async/tile).
// ============================================================
template<bool OBF16, bool BIAS, bool GELU, bool RES>
__global__ __launch_bounds__(256)
void gemm_wmma_k(const __bf16* __restrict__ A0, const __bf16* __restrict__ Bt,
                 void* __restrict__ Cv, const float* __restrict__ bias,
                 const float* __restrict__ resid,
                 int M, int N, int K, int tilesN, int ldc,
                 long long aBatch, long long bBatch,
                 int cDiv, long long cOuter, long long cInner)
{
  __shared__ __bf16 sA[2][TM * ROWP];
  __shared__ __bf16 sB[2][TN * ROWP];

  const int tid  = threadIdx.x;
  const int bt   = blockIdx.y;
  const int tMi  = blockIdx.x / tilesN;
  const int tNi  = blockIdx.x % tilesN;
  const int lane = tid & 31;
  const int wid  = tid >> 5;
  const int hi   = (lane >> 4) & 1;

  const __bf16* A = A0 + (long long)bt * aBatch;
  const __bf16* B = Bt + (long long)bt * bBatch;
  const long long cOff = (long long)(bt / cDiv) * cOuter + (long long)(bt % cDiv) * cInner;

  v8f acc[4];
  #pragma unroll
  for (int j = 0; j < 4; ++j)
    #pragma unroll
    for (int e = 0; e < 8; ++e) acc[j][e] = 0.0f;

  const int rA = wid * 16 + (lane & 15);
  const int nk = K / TK;

  // Each thread issues exactly 3 async b128 loads per tile (2 for A, 1 for B);
  // OOB rows clamp the source row (their C rows are discarded at the epilogue),
  // keeping per-wave ASYNCcnt arithmetic exact and issue counts uniform.
  auto issue_tile = [&](int it, int pb) {
    int k0 = it * TK;
    #pragma unroll
    for (int i = 0; i < 2; ++i) {
      int c    = tid + 256 * i;          // 512 chunks of 8 bf16
      int row  = c >> 2;
      int col  = (c & 3) * 8;
      int gm   = tMi * TM + row;
      int gsrc = (gm < M) ? gm : (M - 1);
      const __bf16* gp = A + (long long)gsrc * K + k0 + col;
      unsigned lp = (unsigned)(unsigned long long)&sA[pb][row * ROWP + col];
      asm volatile("global_load_async_to_lds_b128 %0, %1, off"
                   :: "v"(lp), "v"(gp) : "memory");
    }
    {
      int row = tid >> 2;
      int col = (tid & 3) * 8;
      const __bf16* gp = B + (long long)(tNi * TN + row) * K + k0 + col;
      unsigned lp = (unsigned)(unsigned long long)&sB[pb][row * ROWP + col];
      asm volatile("global_load_async_to_lds_b128 %0, %1, off"
                   :: "v"(lp), "v"(gp) : "memory");
    }
  };

  issue_tile(0, 0);
  for (int it = 0; it < nk; ++it) {
    int cur = it & 1;
    if (it + 1 < nk) {
      issue_tile(it + 1, cur ^ 1);
      // previous tile's 3 loads done when <=3 outstanding (in-order per wave)
      asm volatile("s_wait_asynccnt 3" ::: "memory");
    } else {
      asm volatile("s_wait_asynccnt 0" ::: "memory");
    }
    __syncthreads();               // tile `it` visible to all waves

    // ---- fragments per ISA 16-bit A/B VGPR layout; 4 back-to-back WMMAs ----
    const __bf16* pA = &sA[cur][0];
    const __bf16* pB = &sB[cur][0];
    v8u au;
    #pragma unroll
    for (int r = 0; r < 8; ++r) {
      int kb = (r >> 2) * 16 + hi * 8 + (r & 3) * 2;
      au[r] = *(const unsigned*)&pA[rA * ROWP + kb];
    }
    v8u bu[4];
    #pragma unroll
    for (int j = 0; j < 4; ++j) {
      int rB = j * 16 + (lane & 15);
      #pragma unroll
      for (int r = 0; r < 8; ++r) {
        int kb = (r >> 2) * 16 + hi * 8 + (r & 3) * 2;
        bu[j][r] = *(const unsigned*)&pB[rB * ROWP + kb];
      }
    }
    v16bf afrag = __builtin_bit_cast(v16bf, au);
    #pragma unroll
    for (int j = 0; j < 4; ++j) {
      v16bf bfrag = __builtin_bit_cast(v16bf, bu[j]);
      acc[j] = __builtin_amdgcn_wmma_f32_16x16x32_bf16(
                 false, afrag, false, bfrag, (short)0, acc[j], false, false);
    }
    __syncthreads();               // all waves done reading before buffer reuse
  }

  // ---- epilogue (C layout: VGPR r -> M = r + 8*hi, lane&15 -> N) ----
  #pragma unroll
  for (int j = 0; j < 4; ++j) {
    #pragma unroll
    for (int r = 0; r < 8; ++r) {
      int m = tMi * TM + wid * 16 + r + hi * 8;
      int n = tNi * TN + j * 16 + (lane & 15);
      if (m < M) {
        float v = acc[j][r];
        if (BIAS) v += bias[n];
        if (GELU) v = 0.5f * v * (1.0f + erff(v * 0.70710678118654752f));
        long long off = cOff + (long long)m * ldc + n;
        if (RES) v += resid[off];
        if (OBF16) ((__bf16*)Cv)[off] = (__bf16)v;
        else       ((float*)Cv)[off]  = v;
      }
    }
  }
}

// ============================================================
// LayerNorm (C=768) -> bf16
// ============================================================
__global__ __launch_bounds__(256)
void ln_k(const float* __restrict__ x, const float* __restrict__ g,
          const float* __restrict__ b, __bf16* __restrict__ y)
{
  __shared__ float s1[256], s2[256];
  const int row = blockIdx.x;
  const int t   = threadIdx.x;
  const float* xr = x + (long long)row * CC;
  float v[3], sum = 0.f, sq = 0.f;
  #pragma unroll
  for (int i = 0; i < 3; ++i) { v[i] = xr[t + 256 * i]; sum += v[i]; sq += v[i] * v[i]; }
  s1[t] = sum; s2[t] = sq; __syncthreads();
  for (int o = 128; o > 0; o >>= 1) {
    if (t < o) { s1[t] += s1[t + o]; s2[t] += s2[t + o]; }
    __syncthreads();
  }
  float mu = s1[0] * (1.0f / CC);
  float var = s2[0] * (1.0f / CC) - mu * mu;
  float rs = rsqrtf(var + 1e-5f);
  #pragma unroll
  for (int i = 0; i < 3; ++i) {
    int c = t + 256 * i;
    y[(long long)row * CC + c] = (__bf16)((v[i] - mu) * rs * g[c] + b[c]);
  }
}

// ============================================================
// Softmax on fp32 rows of 320 (one wave32 per row), pre-scale 1/8.
// Writes fp32 in place (attn output) + bf16 copy for the attn@V GEMM.
// ============================================================
__global__ __launch_bounds__(256)
void softmax_k(float* __restrict__ a, __bf16* __restrict__ abf, int rows)
{
  int wrow = blockIdx.x * 8 + (threadIdx.x >> 5);
  int lane = threadIdx.x & 31;
  if (wrow >= rows) return;
  float*  r  = a   + (long long)wrow * LL;
  __bf16* rb = abf + (long long)wrow * LL;
  float v[10], mx = -3.0e38f;
  #pragma unroll
  for (int i = 0; i < 10; ++i) { v[i] = r[lane + 32 * i] * 0.125f; mx = fmaxf(mx, v[i]); }
  #pragma unroll
  for (int o = 16; o > 0; o >>= 1) mx = fmaxf(mx, __shfl_xor(mx, o, 32));
  float s = 0.f;
  #pragma unroll
  for (int i = 0; i < 10; ++i) { v[i] = __expf(v[i] - mx); s += v[i]; }
  #pragma unroll
  for (int o = 16; o > 0; o >>= 1) s += __shfl_xor(s, o, 32);
  float inv = 1.0f / s;
  #pragma unroll
  for (int i = 0; i < 10; ++i) {
    float p = v[i] * inv;
    r[lane + 32 * i]  = p;
    rb[lane + 32 * i] = (__bf16)p;
  }
}

// ============================================================
// Split qkv [B,L,3C] bf16 -> q[B,H,L,64], k[B,H,L,64], vT[B,H,64,L]
// ============================================================
__global__ void repack_k(const __bf16* __restrict__ qkv, __bf16* __restrict__ q,
                         __bf16* __restrict__ k, __bf16* __restrict__ vT)
{
  long long idx = (long long)blockIdx.x * 256 + threadIdx.x; // B*H*L*64
  if (idx >= (long long)SB * HH * LL * HD) return;
  int d = idx & 63;
  long long t = idx >> 6;
  int l = t % LL; t /= LL;
  int h = t % HH;
  int b = t / HH;
  long long base = ((long long)(b * LL + l)) * (3 * CC) + h * HD + d;
  q[idx] = qkv[base];
  k[idx] = qkv[base + CC];
  vT[(((long long)(b * HH + h)) * HD + d) * LL + l] = qkv[base + 2 * CC];
}

// ============================================================
// attn_t[b][j] = mean over (h,q<64) of attn[b][h][q][64+j]
// ============================================================
__global__ __launch_bounds__(256)
void scorered_k(const float* __restrict__ attn, float* __restrict__ scores)
{
  int b = blockIdx.x, j = threadIdx.x;
  const float* p = attn + (long long)b * HH * LL * LL + LT + j;
  float s = 0.f;
  for (int h = 0; h < HH; ++h)
    for (int q = 0; q < LT; ++q)
      s += p[(long long)h * LL * LL + q * LL];
  scores[b * LS + j] = s * (1.0f / (HH * LT));
}

// ============================================================
// Per-batch descending bitonic argsort of 256 scores (stable tie-break),
// emit keep/removed global indices (as float) + topk local indices.
// ============================================================
__global__ __launch_bounds__(256)
void sort_k(const float* __restrict__ scores, const int* __restrict__ gis,
            float* __restrict__ keep, float* __restrict__ rem, int* __restrict__ topk)
{
  __shared__ float sc[256];
  __shared__ int   si[256];
  int b = blockIdx.x, t = threadIdx.x;
  sc[t] = scores[b * LS + t];
  si[t] = t;
  __syncthreads();
  for (int k = 2; k <= 256; k <<= 1) {
    for (int j = k >> 1; j > 0; j >>= 1) {
      int ixj = t ^ j;
      if (ixj > t) {
        bool inc = ((t & k) == 0);
        float a = sc[t], bb = sc[ixj];
        int ia = si[t], ib = si[ixj];
        bool before = (bb > a) || (bb == a && ib < ia); // partner should precede
        if (before == inc) { sc[t] = bb; sc[ixj] = a; si[t] = ib; si[ixj] = ia; }
      }
      __syncthreads();
    }
  }
  int o = si[t];
  if (t < KEEPN) {
    keep[b * KEEPN + t] = (float)gis[b * LS + o];
    topk[b * KEEPN + t] = o;
  } else {
    rem[b * REMN + (t - KEEPN)] = (float)gis[b * LS + o];
  }
}

// ============================================================
// Build new token stream [B,244,768] from x1 [B,320,768] via topk
// ============================================================
__global__ void gather_k(const float* __restrict__ x1, const int* __restrict__ topk,
                         float* __restrict__ out)
{
  long long idx = (long long)blockIdx.x * 256 + threadIdx.x;
  if (idx >= (long long)SB * LNEW * CC) return;
  int c = idx % CC;
  long long t = idx / CC;
  int l = t % LNEW;
  int b = t / LNEW;
  int srcl = (l < LT) ? l : (LT + topk[b * KEEPN + (l - LT)]);
  out[idx] = x1[((long long)b * LL + srcl) * CC + c];
}

__global__ void i2f_k(const int* __restrict__ s, float* __restrict__ d, int n)
{
  int i = blockIdx.x * 256 + threadIdx.x;
  if (i < n) d[i] = (float)s[i];
}

// Weight fp32 [K][N] -> bf16 transposed [N][K]
__global__ void wtrans_k(const float* __restrict__ W, __bf16* __restrict__ Wt,
                         int K, int N)
{
  long long idx = (long long)blockIdx.x * 256 + threadIdx.x;
  if (idx >= (long long)K * N) return;
  int k = idx % K;
  long long n = idx / K;
  Wt[idx] = (__bf16)W[(long long)k * N + n];
}

// ============================================================
extern "C" void kernel_launch(void* const* d_in, const int* in_sizes, int n_in,
                              void* d_out, int out_size, void* d_ws, size_t ws_size,
                              hipStream_t stream)
{
  (void)in_sizes; (void)n_in; (void)out_size; (void)ws_size;
  const float* xs[2]   = { (const float*)d_in[0], (const float*)d_in[1] };
  const int*   git[2]  = { (const int*)d_in[2],   (const int*)d_in[3]   };
  const int*   gis[2]  = { (const int*)d_in[4],   (const int*)d_in[5]   };
  const float* ln1g = (const float*)d_in[6];
  const float* ln1b = (const float*)d_in[7];
  const float* wqkv = (const float*)d_in[8];
  const float* wprj = (const float*)d_in[9];
  const float* bprj = (const float*)d_in[10];
  const float* ln2g = (const float*)d_in[11];
  const float* ln2b = (const float*)d_in[12];
  const float* w1   = (const float*)d_in[13];
  const float* b1   = (const float*)d_in[14];
  const float* w2   = (const float*)d_in[15];
  const float* b2   = (const float*)d_in[16];

  // ---- workspace arena ----
  char* ws = (char*)d_ws;
  size_t cur = 0;
  auto alloc = [&](size_t bytes) { size_t r = cur; cur += (bytes + 255) & ~(size_t)255; return r; };
  __bf16* wqkvt  = (__bf16*)(ws + alloc((size_t)CC * 3 * CC * 2));
  __bf16* wprjt  = (__bf16*)(ws + alloc((size_t)CC * CC * 2));
  __bf16* w1t    = (__bf16*)(ws + alloc((size_t)CC * 4 * CC * 2));
  __bf16* w2t    = (__bf16*)(ws + alloc((size_t)4 * CC * CC * 2));
  size_t  lnoff  = alloc((size_t)SB * LL * CC * 2);                 // lnbuf region
  __bf16* lnbuf  = (__bf16*)(ws + lnoff);                           // xln / ln2 (reused)
  __bf16* bigbuf = (__bf16*)(ws + alloc((size_t)SB * LNEW * 4 * CC * 2)); // qkv / mlp-hidden
  __bf16* qbuf   = (__bf16*)(ws + alloc((size_t)SB * HH * LL * HD * 2));
  __bf16* kbuf   = (__bf16*)(ws + alloc((size_t)SB * HH * LL * HD * 2));
  __bf16* vTbuf  = (__bf16*)(ws + alloc((size_t)SB * HH * HD * LL * 2));
  __bf16* aout   = (__bf16*)(ws + alloc((size_t)SB * LL * CC * 2));        // attention out
  float*  x1     = (float*) (ws + alloc((size_t)SB * LL * CC * 4));        // post-attn residual
  float*  scores = (float*) (ws + alloc((size_t)SB * LS * 4));
  int*    topk   = (int*)   (ws + alloc((size_t)SB * KEEPN * 4));
  // bf16 attention map (78.6MB) aliases [lnbuf .. bigbuf .. qbuf] (79.4MB),
  // all of which are dead between the scores GEMM and ln2/fc1.
  __bf16* attnbf = (__bf16*)(ws + lnoff);

  // ---- one-time weight prep (deterministic, re-run each launch) ----
  {
    long long n;
    n = (long long)CC * 3 * CC; wtrans_k<<<(n + 255) / 256, 256, 0, stream>>>(wqkv, wqkvt, CC, 3 * CC);
    n = (long long)CC * CC;     wtrans_k<<<(n + 255) / 256, 256, 0, stream>>>(wprj, wprjt, CC, CC);
    n = (long long)CC * 4 * CC; wtrans_k<<<(n + 255) / 256, 256, 0, stream>>>(w1,   w1t,   CC, 4 * CC);
    n = (long long)4 * CC * CC; wtrans_k<<<(n + 255) / 256, 256, 0, stream>>>(w2,   w2t,   4 * CC, CC);
  }

  const long long STREAM_STRIDE = 45328384ll; // per-stream output floats
  for (int s = 0; s < 2; ++s) {
    float* ob     = (float*)d_out + (long long)s * STREAM_STRIDE;
    float* o_x    = ob;                 // [32,244,768]
    float* o_git  = ob + 5996544;       // [32,64]
    float* o_keep = ob + 5998592;       // [32,180]
    float* o_rem  = ob + 6004352;       // [32,76]
    float* o_attn = ob + 6006784;       // [32,12,320,320]
    const float* xin = xs[s];

    // 1) LN1 -> bf16
    ln_k<<<SB * LL, 256, 0, stream>>>(xin, ln1g, ln1b, lnbuf);

    // 2) QKV GEMM: [10240x768]x[768x2304] -> bf16 qkv
    {
      int M = SB * LL, N = 3 * CC, K = CC, tN = N / TN, tM = (M + TM - 1) / TM;
      gemm_wmma_k<true, false, false, false><<<dim3(tM * tN, 1), 256, 0, stream>>>(
          lnbuf, wqkvt, bigbuf, nullptr, nullptr, M, N, K, tN, N, 0, 0, 1, 0, 0);
    }

    // 3) split heads (+ v transposed)
    {
      long long n = (long long)SB * HH * LL * HD;
      repack_k<<<(n + 255) / 256, 256, 0, stream>>>(bigbuf, qbuf, kbuf, vTbuf);
    }

    // 4) scores = q k^T (batched over B*H), raw fp32 into attn output region
    {
      int M = LL, N = LL, K = HD, tN = N / TN, tM = (M + TM - 1) / TM;
      gemm_wmma_k<false, false, false, false><<<dim3(tM * tN, SB * HH), 256, 0, stream>>>(
          qbuf, kbuf, o_attn, nullptr, nullptr, M, N, K, tN, LL,
          (long long)LL * HD, (long long)LL * HD, 1, (long long)LL * LL, 0);
    }

    // 5) softmax (scale 1/8): fp32 attn output + bf16 copy for attn@V
    {
      int rows = SB * HH * LL;
      softmax_k<<<(rows + 7) / 8, 256, 0, stream>>>(o_attn, attnbf, rows);
    }

    // 6) out = attn @ v  (bf16 async path; scatter heads into [B,L,C] bf16)
    {
      int M = LL, N = HD, K = LL, tN = N / TN, tM = (M + TM - 1) / TM;
      gemm_wmma_k<true, false, false, false><<<dim3(tM * tN, SB * HH), 256, 0, stream>>>(
          attnbf, vTbuf, aout, nullptr, nullptr, M, N, K, tN, CC,
          (long long)LL * LL, (long long)HD * LL, HH, (long long)LL * CC, (long long)HD);
    }

    // 7) proj + bias + residual(x) -> x1 fp32
    {
      int M = SB * LL, N = CC, K = CC, tN = N / TN, tM = (M + TM - 1) / TM;
      gemm_wmma_k<false, true, false, true><<<dim3(tM * tN, 1), 256, 0, stream>>>(
          aout, wprjt, x1, bprj, xin, M, N, K, tN, CC, 0, 0, 1, 0, 0);
    }

    // 8-10) candidate elimination
    scorered_k<<<SB, 256, 0, stream>>>(o_attn, scores);
    sort_k<<<SB, 256, 0, stream>>>(scores, gis[s], o_keep, o_rem, topk);
    {
      long long n = (long long)SB * LNEW * CC;
      gather_k<<<(n + 255) / 256, 256, 0, stream>>>(x1, topk, o_x);
    }
    i2f_k<<<(SB * LT + 255) / 256, 256, 0, stream>>>(git[s], o_git, SB * LT);

    // 11) LN2 -> bf16 (reclaims lnbuf region; attnbf dead by now)
    ln_k<<<SB * LNEW, 256, 0, stream>>>(o_x, ln2g, ln2b, lnbuf);

    // 12) fc1 + bias + GELU -> bf16 hidden
    {
      int M = SB * LNEW, N = 4 * CC, K = CC, tN = N / TN, tM = (M + TM - 1) / TM;
      gemm_wmma_k<true, true, true, false><<<dim3(tM * tN, 1), 256, 0, stream>>>(
          lnbuf, w1t, bigbuf, b1, nullptr, M, N, K, tN, N, 0, 0, 1, 0, 0);
    }

    // 13) fc2 + bias + residual (in place on o_x)
    {
      int M = SB * LNEW, N = CC, K = 4 * CC, tN = N / TN, tM = (M + TM - 1) / TM;
      gemm_wmma_k<false, true, false, true><<<dim3(tM * tN, 1), 256, 0, stream>>>(
          bigbuf, w2t, o_x, b2, o_x, M, N, K, tN, CC, 0, 0, 1, 0, 0);
    }
  }
}
